// MHSA_5884105195621
// MI455X (gfx1250) — matrix operand order repaired and checked by reading, not compiled, not run on hardware
//
#include <hip/hip_runtime.h>

// ---------------------------------------------------------------------------
// MHSA (dual-attention w/ CNN positional bias) for MI455X / gfx1250.
// bf16 WMMA (v_wmma_f32_16x16x32_bf16) for all GEMMs, flash-attention fusion
// so the 4096x4096 attention matrix never touches HBM. K/V tiles are staged
// into LDS with GLOBAL_LOAD_ASYNC_TO_LDS_B128 (ASYNCcnt) — the CDNA5 async
// global->LDS path.
// ---------------------------------------------------------------------------

#define BDIM 8
#define CDIM 128
#define NPIX 4096          // 64*64

typedef __attribute__((ext_vector_type(16))) __bf16        bf16x16;
typedef __attribute__((ext_vector_type(8)))  float         f32x8;
typedef __attribute__((ext_vector_type(4)))  unsigned int  u32x4;

union FragU { bf16x16 v; u32x4 q[2]; };

// CDNA5 async global->LDS copy (16B per lane), tracked by ASYNCcnt.
__device__ __forceinline__ void async_g2l_b128(void* lds, const void* gaddr) {
  unsigned lds_off = (unsigned)(size_t)lds;    // low 32 bits of generic = LDS byte addr
  asm volatile("global_load_async_to_lds_b128 %0, %1, off"
               :: "v"(lds_off), "v"(gaddr) : "memory");
}
__device__ __forceinline__ void wait_async_all() {
  asm volatile("s_wait_asynccnt 0" ::: "memory");
}

// ===================== small prep kernels ==================================

// per-(b,c) mean & max over H*W of x2
__global__ __launch_bounds__(256) void poolbc_kernel(const float* __restrict__ x2,
                                                     float* __restrict__ av,
                                                     float* __restrict__ mx) {
  __shared__ float ssum[256], smax[256];
  int bc = blockIdx.x, tid = threadIdx.x;
  const float* p = x2 + (size_t)bc * NPIX;
  float s = 0.f, m = -1e30f;
  for (int i = tid; i < NPIX; i += 256) { float v = p[i]; s += v; m = fmaxf(m, v); }
  ssum[tid] = s; smax[tid] = m; __syncthreads();
  for (int o = 128; o > 0; o >>= 1) {
    if (tid < o) { ssum[tid] += ssum[tid + o]; smax[tid] = fmaxf(smax[tid], smax[tid + o]); }
    __syncthreads();
  }
  if (tid == 0) { av[bc] = ssum[0] * (1.f / NPIX); mx[bc] = smax[0]; }
}

// conv1d(2->1, k=5, pad=2) over channel axis -> ch[b,c]
__global__ void chconv_kernel(const float* __restrict__ av, const float* __restrict__ mx,
                              const float* __restrict__ c1w, float* __restrict__ ch) {
  int i = blockIdx.x * 256 + threadIdx.x;   // b*128 + c
  int b = i >> 7, c = i & 127;
  float acc = 0.f;
  for (int t = 0; t < 5; ++t) {
    int cc = c + t - 2;
    if (cc >= 0 && cc < CDIM)
      acc += c1w[t] * av[b * CDIM + cc] + c1w[5 + t] * mx[b * CDIM + cc];
  }
  ch[i] = acc;
}

// per-(b,n) mean & max over channels of x2
__global__ __launch_bounds__(256) void poolc_kernel(const float* __restrict__ x2,
                                                    float* __restrict__ cmean,
                                                    float* __restrict__ cmax) {
  int i = blockIdx.x * 256 + threadIdx.x;   // b*4096 + n
  int b = i >> 12, n = i & 4095;
  const float* p = x2 + (size_t)b * CDIM * NPIX + n;
  float s = 0.f, m = -1e30f;
  for (int c = 0; c < CDIM; ++c) { float v = p[(size_t)c * NPIX]; s += v; m = fmaxf(m, v); }
  cmean[i] = s * (1.f / CDIM); cmax[i] = m;
}

// conv2d(2->1, 7x7, pad=3) + bias -> spmap[b,n]
__global__ __launch_bounds__(256) void spconv_kernel(const float* __restrict__ cmean,
                                                     const float* __restrict__ cmax,
                                                     const float* __restrict__ spw,
                                                     const float* __restrict__ spb,
                                                     float* __restrict__ spmap) {
  int i = blockIdx.x * 256 + threadIdx.x;
  int b = i >> 12, n = i & 4095, h = n >> 6, wx = n & 63;
  float acc = spb[0];
  for (int dy = 0; dy < 7; ++dy) {
    int y = h + dy - 3; if ((unsigned)y >= 64u) continue;
    for (int dx = 0; dx < 7; ++dx) {
      int x = wx + dx - 3; if ((unsigned)x >= 64u) continue;
      int nn = b * NPIX + y * 64 + x;
      acc += spw[dy * 7 + dx] * cmean[nn] + spw[49 + dy * 7 + dx] * cmax[nn];
    }
  }
  spmap[i] = acc;
}

// Avec[b,o] = ck_b[o] + sum_c ck_w[o,c]*ch[b,c];  svec[o] = sum_c ck_w[o,c]
__global__ void avec_kernel(const float* __restrict__ ckw, const float* __restrict__ ckb,
                            const float* __restrict__ ch, float* __restrict__ Avec,
                            float* __restrict__ svec) {
  int i = blockIdx.x * 256 + threadIdx.x;   // b*128 + o
  int b = i >> 7, o = i & 127;
  float a = ckb[o], s = 0.f;
  for (int c = 0; c < CDIM; ++c) { float wv = ckw[o * CDIM + c]; a += wv * ch[b * CDIM + c]; s += wv; }
  Avec[i] = a;
  if (b == 0) svec[o] = s;
}

// convert q_w,k_w,v_w (f32, [o][c] row-major) -> packed bf16 [3][128][128]
__global__ void wconv_kernel(const float* __restrict__ qw, const float* __restrict__ kw,
                             const float* __restrict__ vw, __bf16* __restrict__ Wb) {
  int i = blockIdx.x * 256 + threadIdx.x;   // 0..49151
  const float* src = (i < 16384) ? qw : ((i < 32768) ? kw : vw);
  Wb[i] = (__bf16)src[i & 16383];
}

// ===================== projection kernel (WMMA) ============================
// Q/K -> Qt/Kt[b][n][c] bf16 (row n contiguous over c)
// V   -> Vt[b][c][m]     bf16 (row c contiguous over m)
__global__ __launch_bounds__(256) void proj_kernel(
    const float* __restrict__ x, const __bf16* __restrict__ Wb,
    const float* __restrict__ qb, const float* __restrict__ kbias,
    const float* __restrict__ vb,
    __bf16* __restrict__ Qt, __bf16* __restrict__ Kt, __bf16* __restrict__ Vt) {
  __shared__ __bf16 xT[128 * 136];          // x^T tile [n][c], padded stride

  const int b  = blockIdx.x >> 5;
  const int n0 = (blockIdx.x & 31) * 128;
  const int tid = threadIdx.x;
  const int wv = tid >> 5, lane = tid & 31, l16 = lane & 15, lh = lane >> 4;

  // load x tile [c=128][n=128] f32, transpose+convert into LDS
  const float* xg = x + (size_t)b * CDIM * NPIX + n0;
  for (int it = 0; it < 16; ++it) {
    int chunk = tid + it * 256;             // 0..4095 : c = chunk>>5, f4 = chunk&31
    int c = chunk >> 5, f4 = chunk & 31;
    const float* src = xg + (size_t)c * NPIX + f4 * 4;
    int n = f4 * 4;
    xT[(n + 0) * 136 + c] = (__bf16)src[0];
    xT[(n + 1) * 136 + c] = (__bf16)src[1];
    xT[(n + 2) * 136 + c] = (__bf16)src[2];
    xT[(n + 3) * 136 + c] = (__bf16)src[3];
  }
  __syncthreads();

  // A fragments (x^T rows, shared by all three projections)
  FragU afrag[4];
  const __bf16* xrow = &xT[(wv * 16 + l16) * 136];
  for (int kc = 0; kc < 4; ++kc) {
    afrag[kc].q[0] = *(const u32x4*)(xrow + kc * 32 + lh * 8);
    afrag[kc].q[1] = *(const u32x4*)(xrow + kc * 32 + 16 + lh * 8);
  }

  const f32x8 fzero = {};
  for (int p = 0; p < 3; ++p) {
    const __bf16* W = Wb + p * 16384;
    const float* bias = (p == 0) ? qb : ((p == 1) ? kbias : vb);
    f32x8 acc[8];
    for (int t = 0; t < 8; ++t) acc[t] = fzero;
    for (int kc = 0; kc < 4; ++kc) {
      for (int t = 0; t < 8; ++t) {
        FragU bb;
        const __bf16* wrow = W + (size_t)(t * 16 + l16) * CDIM + kc * 32 + lh * 8;
        bb.q[0] = *(const u32x4*)wrow;
        bb.q[1] = *(const u32x4*)(wrow + 16);
        acc[t] = __builtin_amdgcn_wmma_f32_16x16x32_bf16(
            false, afrag[kc].v, false, bb.v, (short)0, acc[t], false, false);
      }
    }
    for (int t = 0; t < 8; ++t) {
      float bv = bias[t * 16 + l16];
      int n = n0 + wv * 16;
      if (p < 2) {
        __bf16* dst = ((p == 0) ? Qt : Kt) + ((size_t)b * NPIX + n) * CDIM + t * 16 + l16;
        for (int r = 0; r < 8; ++r) dst[(size_t)(r + 8 * lh) * CDIM] = (__bf16)(acc[t][r] + bv);
      } else {
        __bf16* dst = Vt + (size_t)b * CDIM * NPIX + (size_t)(t * 16 + l16) * NPIX + n;
        for (int r = 0; r < 8; ++r) dst[r + 8 * lh] = (__bf16)(acc[t][r] + bv);
      }
    }
  }
}

// u[m] = sum_c Avec[c]*q[c,m];  w[m] = sum_c svec[c]*q[c,m]
__global__ __launch_bounds__(256) void uw_kernel(const __bf16* __restrict__ Qt,
                                                 const float* __restrict__ Avec,
                                                 const float* __restrict__ svec,
                                                 float* __restrict__ u, float* __restrict__ w) {
  int i = blockIdx.x * 256 + threadIdx.x;   // b*4096 + m
  int b = i >> 12;
  const __bf16* q = Qt + (size_t)i * CDIM;
  float su = 0.f, sw = 0.f;
  for (int c = 0; c < CDIM; ++c) {
    float qv = (float)q[c];
    su += Avec[b * CDIM + c] * qv;
    sw += svec[c] * qv;
  }
  u[i] = su; w[i] = sw;
}

// ===================== flash attention (WMMA + async LDS) ==================
// energy[n,m] = q.k + u[m] + sp[n]*w[m]; softmax over m streamed online.
__global__ __launch_bounds__(256) void attn_kernel(
    const __bf16* __restrict__ Qt, const __bf16* __restrict__ Kt,
    const __bf16* __restrict__ Vt, const float* __restrict__ u,
    const float* __restrict__ w, const float* __restrict__ spmap,
    float* __restrict__ out) {
  __shared__ __bf16 kps[9216];              // K tile [64][136]  UNION  P stage [8][16][72]
  __shared__ __bf16 vs[128 * 72];           // V^T tile [c][m], padded stride 72

  const int b  = blockIdx.x >> 5;
  const int n0 = (blockIdx.x & 31) * 128;
  const int tid = threadIdx.x;
  const int wv = tid >> 5, lane = tid & 31, l16 = lane & 15, lh = lane >> 4;

  // Q fragments for this wave's 16 rows — kept in registers for all 64 m-tiles
  FragU qf[4];
  {
    const __bf16* qrow = Qt + ((size_t)b * NPIX + n0 + wv * 16 + l16) * CDIM;
    for (int kc = 0; kc < 4; ++kc) {
      qf[kc].q[0] = *(const u32x4*)(qrow + kc * 32 + lh * 8);
      qf[kc].q[1] = *(const u32x4*)(qrow + kc * 32 + 16 + lh * 8);
    }
  }

  float spn[8], mrow[8], lrow[8];
  for (int r = 0; r < 8; ++r) {
    spn[r] = spmap[b * NPIX + n0 + wv * 16 + r + 8 * lh];
    mrow[r] = -1e30f; lrow[r] = 0.f;
  }
  const f32x8 fzero = {};
  f32x8 oacc[8];
  for (int t = 0; t < 8; ++t) oacc[t] = fzero;

  for (int m0 = 0; m0 < NPIX; m0 += 64) {
    __syncthreads();                        // prev iter done with kps (P) and vs
    // stage K tile [64 rows m][128 c] and V^T tile [128 rows c][64 m]
    // via CDNA5 async global->LDS DMA (no VGPR round-trip, ASYNCcnt-tracked)
    const __bf16* kg = Kt + ((size_t)b * NPIX + m0) * CDIM;
    for (int it = 0; it < 4; ++it) {
      int chunk = tid + it * 256;           // 0..1023
      int row = chunk >> 4, col8 = chunk & 15;
      async_g2l_b128(&kps[row * 136 + col8 * 8], kg + (size_t)chunk * 8);
    }
    const __bf16* vg = Vt + (size_t)b * CDIM * NPIX + m0;
    for (int it = 0; it < 4; ++it) {
      int chunk = tid + it * 256;           // 128 rows x 8 chunks
      int row = chunk >> 3, col8 = chunk & 7;
      async_g2l_b128(&vs[row * 72 + col8 * 8], vg + (size_t)row * NPIX + col8 * 8);
    }
    wait_async_all();                       // s_wait_asynccnt 0
    __syncthreads();

    // ---- S = Q.K^T : 4 column tiles of 16, K-dim = 128 ----
    f32x8 s[4];
    for (int t = 0; t < 4; ++t) s[t] = fzero;
    for (int kc = 0; kc < 4; ++kc) {
      for (int t = 0; t < 4; ++t) {
        FragU bb;
        const __bf16* krow = &kps[(t * 16 + l16) * 136 + kc * 32 + lh * 8];
        bb.q[0] = *(const u32x4*)krow;
        bb.q[1] = *(const u32x4*)(krow + 16);
        s[t] = __builtin_amdgcn_wmma_f32_16x16x32_bf16(
            false, qf[kc].v, false, bb.v, (short)0, s[t], false, false);
      }
    }
    __syncthreads();                        // everyone done reading K tile (kps reused for P)

    // ---- bias + online softmax (rows live in 16-lane groups) ----
    float ut[4], wt[4];
    for (int t = 0; t < 4; ++t) {
      ut[t] = u[b * NPIX + m0 + t * 16 + l16];
      wt[t] = w[b * NPIX + m0 + t * 16 + l16];
    }
    float newmax[8];
    for (int r = 0; r < 8; ++r) {
      float mxv = -1e30f;
      for (int t = 0; t < 4; ++t) {
        float e = s[t][r] + ut[t] + spn[r] * wt[t];
        s[t][r] = e;
        mxv = fmaxf(mxv, e);
      }
      for (int off = 1; off < 16; off <<= 1) mxv = fmaxf(mxv, __shfl_xor(mxv, off));
      newmax[r] = fmaxf(mrow[r], mxv);
    }
    float rs[8];
    for (int r = 0; r < 8; ++r) rs[r] = 0.f;
    for (int t = 0; t < 4; ++t)
      for (int r = 0; r < 8; ++r) {
        float p = __expf(s[t][r] - newmax[r]);
        s[t][r] = p; rs[r] += p;
      }
    for (int r = 0; r < 8; ++r) {
      for (int off = 1; off < 16; off <<= 1) rs[r] += __shfl_xor(rs[r], off);
      float corr = __expf(mrow[r] - newmax[r]);
      lrow[r] = lrow[r] * corr + rs[r];
      mrow[r] = newmax[r];
      for (int t = 0; t < 8; ++t) oacc[t][r] *= corr;
    }

    // ---- stage P (bf16) into per-wave LDS region, re-read as A fragments ----
    __bf16* pw = &kps[wv * 16 * 72];
    for (int t = 0; t < 4; ++t)
      for (int r = 0; r < 8; ++r)
        pw[(r + 8 * lh) * 72 + t * 16 + l16] = (__bf16)s[t][r];
    asm volatile("s_wait_dscnt 0" ::: "memory");   // same-wave LDS RAW

    // ---- O += P.V : 8 channel tiles, K-dim = 64 ----
    for (int half = 0; half < 2; ++half) {
      FragU pa;
      const __bf16* prow = pw + l16 * 72 + half * 32;
      pa.q[0] = *(const u32x4*)(prow + lh * 8);
      pa.q[1] = *(const u32x4*)(prow + 16 + lh * 8);
      for (int t = 0; t < 8; ++t) {
        FragU vb;
        const __bf16* vrow = &vs[(t * 16 + l16) * 72 + half * 32 + lh * 8];
        vb.q[0] = *(const u32x4*)vrow;
        vb.q[1] = *(const u32x4*)(vrow + 16);
        oacc[t] = __builtin_amdgcn_wmma_f32_16x16x32_bf16(
            false, pa.v, false, vb.v, (short)0, oacc[t], false, false);
      }
    }
  }

  // ---- normalize & store out[b][c][n] ----
  float inv[8];
  for (int r = 0; r < 8; ++r) inv[r] = 1.f / lrow[r];
  float* og = out + (size_t)b * CDIM * NPIX;
  for (int t = 0; t < 8; ++t)
    for (int r = 0; r < 8; ++r) {
      int c = t * 16 + l16;
      int n = n0 + wv * 16 + r + 8 * lh;
      og[(size_t)c * NPIX + n] = oacc[t][r] * inv[r];
    }
}

// ===================== host launcher =======================================

extern "C" void kernel_launch(void* const* d_in, const int* in_sizes, int n_in,
                              void* d_out, int out_size, void* d_ws, size_t ws_size,
                              hipStream_t stream) {
  const float* x      = (const float*)d_in[0];
  const float* x2     = (const float*)d_in[1];
  const float* q_w    = (const float*)d_in[2];
  const float* q_b    = (const float*)d_in[3];
  const float* k_w    = (const float*)d_in[4];
  const float* k_b    = (const float*)d_in[5];
  const float* v_w    = (const float*)d_in[6];
  const float* v_b    = (const float*)d_in[7];
  const float* ck_w   = (const float*)d_in[8];
  const float* ck_b   = (const float*)d_in[9];
  const float* c1w    = (const float*)d_in[10];
  const float* sp_w   = (const float*)d_in[11];
  const float* sp_b   = (const float*)d_in[12];
  float* out = (float*)d_out;

  char* base = (char*)d_ws;
  size_t off = 0;
  auto take = [&](size_t bytes) -> char* {
    char* p = base + off;
    off = (off + bytes + 255) & ~(size_t)255;
    return p;
  };
  float*  av    = (float*)take(BDIM * CDIM * 4);
  float*  mx    = (float*)take(BDIM * CDIM * 4);
  float*  ch    = (float*)take(BDIM * CDIM * 4);
  float*  cmean = (float*)take((size_t)BDIM * NPIX * 4);
  float*  cmax  = (float*)take((size_t)BDIM * NPIX * 4);
  float*  spmap = (float*)take((size_t)BDIM * NPIX * 4);
  float*  Avec  = (float*)take(BDIM * CDIM * 4);
  float*  svec  = (float*)take(CDIM * 4);
  float*  uvec  = (float*)take((size_t)BDIM * NPIX * 4);
  float*  wvec  = (float*)take((size_t)BDIM * NPIX * 4);
  __bf16* Wb    = (__bf16*)take((size_t)3 * CDIM * CDIM * 2);
  __bf16* Qt    = (__bf16*)take((size_t)BDIM * NPIX * CDIM * 2);
  __bf16* Kt    = (__bf16*)take((size_t)BDIM * NPIX * CDIM * 2);
  __bf16* Vt    = (__bf16*)take((size_t)BDIM * NPIX * CDIM * 2);
  if (off > ws_size) return;   // workspace too small — nothing safe to do

  // CNN bias branch (tiny)
  poolbc_kernel<<<BDIM * CDIM, 256, 0, stream>>>(x2, av, mx);
  poolc_kernel <<<BDIM * NPIX / 256, 256, 0, stream>>>(x2, cmean, cmax);
  chconv_kernel<<<BDIM * CDIM / 256, 256, 0, stream>>>(av, mx, c1w, ch);
  spconv_kernel<<<BDIM * NPIX / 256, 256, 0, stream>>>(cmean, cmax, sp_w, sp_b, spmap);
  avec_kernel  <<<BDIM * CDIM / 256, 256, 0, stream>>>(ck_w, ck_b, ch, Avec, svec);
  wconv_kernel <<<3 * CDIM * CDIM / 256, 256, 0, stream>>>(q_w, k_w, v_w, Wb);

  // projections (WMMA)
  proj_kernel<<<BDIM * (NPIX / 128), 256, 0, stream>>>(x, Wb, q_b, k_b, v_b, Qt, Kt, Vt);
  // rank-1 positional-bias vectors
  uw_kernel<<<BDIM * NPIX / 256, 256, 0, stream>>>(Qt, Avec, svec, uvec, wvec);
  // fused flash attention (WMMA + async LDS staging)
  attn_kernel<<<BDIM * (NPIX / 128), 256, 0, stream>>>(Qt, Kt, Vt, uvec, wvec, spmap, out);
  (void)in_sizes; (void)n_in; (void)out_size;
}
